// QTransformerClassifier_65481071407819
// MI455X (gfx1250) — matrix operand order repaired and checked by reading, not compiled, not run on hardware
//
#include <hip/hip_runtime.h>
#include <hip/hip_bf16.h>
#include <math.h>

// ---------------------------------------------------------------------------
// Problem constants (from reference)
// ---------------------------------------------------------------------------
#define BQ   8
#define SQ   512
#define EQ   128
#define HQ   16
#define DKQ  8
#define LQ   4
#define FQ   512
#define NQQ  4
#define CQ   10
#define BS   (BQ * SQ)     // 4096 token rows
#define LN_EPS 1e-5f

typedef __attribute__((ext_vector_type(16))) _Float16 v16h;
typedef __attribute__((ext_vector_type(8)))  float    v8f;

// ---------------------------------------------------------------------------
// WMMA f16 16x16x32 fragment index maps (CDNA5 ISA 7.12.2, wave32)
// A (16x32, MxK): lanes 0-15 -> M=lane,    halves 0-7 = K 0-7,  halves 8-15 = K 16-23
//                 lanes16-31 -> M=lane-16, halves 0-7 = K 8-15, halves 8-15 = K 24-31
// B (32x16, KxN): lanes 0-15 -> N=lane,    halves 0-7 = K 0-7,  halves 8-15 = K 16-23
//                 lanes16-31 -> N=lane-16, halves 0-7 = K 8-15, halves 8-15 = K 24-31
// C (16x16, f32): VGPR v: lanes 0-15 -> (M=v,   N=lane)
//                         lanes16-31 -> (M=v+8, N=lane-16)
// ---------------------------------------------------------------------------
__device__ __forceinline__ int a_k_of_h(int lane, int h) {
    if (lane < 16) return (h < 8) ? h : (h + 8);
    return (h < 8) ? (h + 8) : (h + 16);
}
__device__ __forceinline__ int b_k_of_h(int lane, int h) {
    return a_k_of_h(lane, h);
}

// ---------------------------------------------------------------------------
// Embedding lookup + sinusoidal positional encoding
// grid = BS blocks, block = 128 threads (one per E element)
// ---------------------------------------------------------------------------
__global__ void qtc_embed_kernel(const int* __restrict__ tokens,
                                 const float* __restrict__ emb,
                                 float* __restrict__ h) {
    int row = blockIdx.x;          // b*S + s
    int e   = threadIdx.x;         // 0..127
    int s   = row & (SQ - 1);
    int tok = tokens[row];
    int i2  = e & ~1;              // 2*(e/2)
    float div = __expf(-(float)i2 * (logf(10000.0f) / (float)EQ));
    float arg = (float)s * div;
    float pe  = (e & 1) ? __cosf(arg) : __sinf(arg);
    h[row * EQ + e] = emb[tok * EQ + e] + pe;
}

// ---------------------------------------------------------------------------
// Ring-measure quantum encoding of one DK=8 row:
//   c_d = cos(x_d + theta_d); out[0] = prod_{j>=1} c_j; out[i] = prod_{j<=i} c_j
// ---------------------------------------------------------------------------
__device__ __forceinline__ void ring_measure_row(const float* x, const float* th,
                                                 float* out) {
    float c[DKQ];
#pragma unroll
    for (int d = 0; d < DKQ; ++d) c[d] = __cosf(x[d] + th[d]);
    float z0 = 1.0f;
#pragma unroll
    for (int d = 1; d < DKQ; ++d) z0 *= c[d];
    out[0] = z0;
    float p = c[0];
#pragma unroll
    for (int d = 1; d < DKQ; ++d) { p *= c[d]; out[d] = p; }
}

// ---------------------------------------------------------------------------
// Flash-style quantum attention: one wave per (b, head, 16-query tile).
// Streams 32 key tiles; scores via v_wmma_f32_16x16x32_f16 (K=8 zero-padded),
// online softmax in registers (row stats are lane-local thanks to C layout),
// P.V via a second WMMA after an LDS repack (C layout -> A layout).
// Scores never touch global memory.
// grid = (S/16, H, B), block = 32 (one wave)
// ---------------------------------------------------------------------------
__global__ void qtc_attn_kernel(const float* __restrict__ h,
                                const float* __restrict__ theta,  // DK for layer
                                float* __restrict__ oattn) {
    __shared__ float encQ[16][DKQ];
    __shared__ float encK[16][DKQ];
    __shared__ float Pt[16][16];

    const int qt   = blockIdx.x;
    const int head = blockIdx.y;
    const int b    = blockIdx.z;
    const int lane = threadIdx.x;
    const int s0   = qt * 16;
    const int half = (lane < 16) ? 0 : 1;
    const int col  = lane & 15;

    // --- encode 16 query rows (lanes 0-15, one row each) ---
    if (lane < 16) {
        float x[DKQ];
        const float* src = h + ((size_t)(b * SQ + s0 + lane)) * EQ + head * DKQ;
#pragma unroll
        for (int d = 0; d < DKQ; ++d) x[d] = src[d];
        ring_measure_row(x, theta, encQ[lane]);
    }
    __syncthreads();

    // --- pack A fragment for queries: 16x32, K 0..7 valid ---
    v16h aQ;
#pragma unroll
    for (int hh = 0; hh < 16; ++hh) {
        int k = a_k_of_h(lane, hh);
        aQ[hh] = (k < DKQ) ? (_Float16)encQ[col][k] : (_Float16)0.0f;
    }

    // --- online softmax state: each lane tracks its 8 rows of the C layout ---
    float m_run[8], l_run[8];
#pragma unroll
    for (int v = 0; v < 8; ++v) { m_run[v] = -INFINITY; l_run[v] = 0.0f; }
    v8f oacc = {};
    const float inv_sqrt_dk = 0.35355339059327373f;

    for (int kt = 0; kt < SQ / 16; ++kt) {
        // encode 16 key rows
        if (lane < 16) {
            float x[DKQ];
            const float* src = h + ((size_t)(b * SQ + kt * 16 + lane)) * EQ + head * DKQ;
#pragma unroll
            for (int d = 0; d < DKQ; ++d) x[d] = src[d];
            ring_measure_row(x, theta, encK[lane]);
        }
        __syncthreads();

        // B fragment: B[k][n] = encK[n][k] (k<8)
        v16h bK;
#pragma unroll
        for (int hh = 0; hh < 16; ++hh) {
            int k = b_k_of_h(lane, hh);
            bK[hh] = (k < DKQ) ? (_Float16)encK[col][k] : (_Float16)0.0f;
        }
        v8f sc = {};
        sc = __builtin_amdgcn_wmma_f32_16x16x32_f16(false, aQ, false, bK,
                                                    (short)0, sc, false, false);

        // online softmax over this 16-key tile
#pragma unroll
        for (int v = 0; v < 8; ++v) {
            float t = sc[v] * inv_sqrt_dk;
            // row max across the 16 lanes of this half-wave
            float tm = t;
#pragma unroll
            for (int off = 1; off < 16; off <<= 1)
                tm = fmaxf(tm, __shfl_xor(tm, off, 16));
            float m_new = fmaxf(m_run[v], tm);
            float alpha = __expf(m_run[v] - m_new);
            float p     = __expf(t - m_new);
            float ps    = p;
#pragma unroll
            for (int off = 1; off < 16; off <<= 1)
                ps += __shfl_xor(ps, off, 16);
            l_run[v] = l_run[v] * alpha + ps;
            m_run[v] = m_new;
            oacc[v] *= alpha;                      // row rescale is lane-local
            Pt[v + half * 8][col] = p;             // C layout -> LDS tile
        }
        __syncthreads();

        // A2 fragment from P (16x16, K pad to 32); B2 = encK (16x8, pad to 32x16)
        v16h a2, b2;
#pragma unroll
        for (int hh = 0; hh < 16; ++hh) {
            int ka = a_k_of_h(lane, hh);
            a2[hh] = (ka < 16) ? (_Float16)Pt[col][ka] : (_Float16)0.0f;
            int kb = b_k_of_h(lane, hh);
            b2[hh] = (kb < 16 && col < DKQ) ? (_Float16)encK[kb][col]
                                            : (_Float16)0.0f;
        }
        oacc = __builtin_amdgcn_wmma_f32_16x16x32_f16(false, a2, false, b2,
                                                      (short)0, oacc, false, false);
        __syncthreads();   // before next tile overwrites encK / Pt
    }

    // epilogue: normalize by l and scatter (col = head-dim, only col<8 valid)
    if (col < DKQ) {
#pragma unroll
        for (int v = 0; v < 8; ++v) {
            int r = v + half * 8;
            oattn[((size_t)(b * SQ + s0 + r)) * EQ + head * DKQ + col] =
                oacc[v] / l_run[v];
        }
    }
}

// ---------------------------------------------------------------------------
// Generic NT GEMM:  C[M x N] = A[M x K] * B[N x K]^T   (f32 in, f16 WMMA, f32 acc)
// One wave per 16x64 output strip: one A fragment is reused across 4 B
// fragments -> 4 back-to-back v_wmma per 32-wide K step (XDL-friendly),
// 4x A-tile reuse. LDS-staged f32 tiles repacked to fragment layout.
// Requires M%16==0, N%64==0, K%32==0.
// grid = (M/16, N/64), block = 32
// ---------------------------------------------------------------------------
#define GEMM_NT 4   // 16-wide N tiles per wave
__global__ void qtc_gemm_nt_wmma(const float* __restrict__ A,
                                 const float* __restrict__ B,
                                 float* __restrict__ C,
                                 int M, int N, int K) {
    __shared__ float As[16][32];
    __shared__ float Bs[GEMM_NT * 16][32];
    const int m0   = blockIdx.x * 16;
    const int n0   = blockIdx.y * (GEMM_NT * 16);
    const int lane = threadIdx.x;
    const int col  = lane & 15;

    v8f acc[GEMM_NT];
#pragma unroll
    for (int t = 0; t < GEMM_NT; ++t) acc[t] = (v8f){};

    for (int kk = 0; kk < K; kk += 32) {
        // cooperative staging of the K chunk (A: 16x32, B: 64x32)
        for (int i = lane; i < 16 * 32; i += 32) {
            int r = i >> 5, c = i & 31;
            As[r][c] = A[(size_t)(m0 + r) * K + kk + c];
        }
        for (int i = lane; i < GEMM_NT * 16 * 32; i += 32) {
            int r = i >> 5, c = i & 31;
            Bs[r][c] = B[(size_t)(n0 + r) * K + kk + c];
        }
        if (kk + 32 < K) {   // hint next K chunk into cache (global_prefetch_b8)
            __builtin_prefetch(&A[(size_t)(m0 + (lane >> 1)) * K + kk + 32], 0, 3);
            __builtin_prefetch(&B[(size_t)(n0 + lane) * K + kk + 32], 0, 3);
        }
        __syncthreads();

        v16h a;
#pragma unroll
        for (int hh = 0; hh < 16; ++hh)
            a[hh] = (_Float16)As[col][a_k_of_h(lane, hh)];

#pragma unroll
        for (int t = 0; t < GEMM_NT; ++t) {
            v16h bfr;
#pragma unroll
            for (int hh = 0; hh < 16; ++hh)
                bfr[hh] = (_Float16)Bs[t * 16 + col][b_k_of_h(lane, hh)];
            acc[t] = __builtin_amdgcn_wmma_f32_16x16x32_f16(
                false, a, false, bfr, (short)0, acc[t], false, false);
        }
        __syncthreads();
    }

    const int mb = m0 + ((lane < 16) ? 0 : 8);
#pragma unroll
    for (int t = 0; t < GEMM_NT; ++t) {
        const int n = n0 + t * 16 + col;
#pragma unroll
        for (int v = 0; v < 8; ++v)
            C[(size_t)(mb + v) * N + n] = acc[t][v];
    }
}

// ---------------------------------------------------------------------------
// Residual add + LayerNorm over E=128; one wave per row, 4 elements per lane.
// grid = BS, block = 32
// ---------------------------------------------------------------------------
__global__ void qtc_add_ln_kernel(float* __restrict__ h,
                                  const float* __restrict__ delta,
                                  const float* __restrict__ g,
                                  const float* __restrict__ bta) {
    const int row  = blockIdx.x;
    const int lane = threadIdx.x;
    float x[4];
    float s = 0.0f;
#pragma unroll
    for (int j = 0; j < 4; ++j) {
        int e = lane * 4 + j;
        x[j]  = h[(size_t)row * EQ + e] + delta[(size_t)row * EQ + e];
        s += x[j];
    }
#pragma unroll
    for (int off = 1; off < 32; off <<= 1) s += __shfl_xor(s, off, 32);
    float mean = s * (1.0f / EQ);
    float vs = 0.0f;
#pragma unroll
    for (int j = 0; j < 4; ++j) { float d = x[j] - mean; vs += d * d; }
#pragma unroll
    for (int off = 1; off < 32; off <<= 1) vs += __shfl_xor(vs, off, 32);
    float rinv = rsqrtf(vs * (1.0f / EQ) + LN_EPS);
#pragma unroll
    for (int j = 0; j < 4; ++j) {
        int e = lane * 4 + j;
        h[(size_t)row * EQ + e] = (x[j] - mean) * rinv * g[e] + bta[e];
    }
}

// ---------------------------------------------------------------------------
// FFN front: a1[row][f] = relu( sum_q cos(h[row][q])*cos(theta_q) * W1[f][q] )
// NQ = 4: scalar (negligible FLOPs). grid*block = BS*F threads.
// ---------------------------------------------------------------------------
__global__ void qtc_ffn1_kernel(const float* __restrict__ h,
                                const float* __restrict__ theta,   // NQ
                                const float* __restrict__ W1,      // F x NQ
                                float* __restrict__ a1) {
    int idx = blockIdx.x * blockDim.x + threadIdx.x;   // row*F + f
    if (idx >= BS * FQ) return;
    int row = idx / FQ, f = idx % FQ;
    float acc = 0.0f;
#pragma unroll
    for (int q = 0; q < NQQ; ++q)
        acc += __cosf(h[(size_t)row * EQ + q]) * __cosf(theta[q]) * W1[f * NQQ + q];
    a1[idx] = fmaxf(acc, 0.0f);
}

// ---------------------------------------------------------------------------
// Mean-pool over S + classifier head (8x10, tiny -> scalar, one block)
// ---------------------------------------------------------------------------
__global__ void qtc_head_kernel(const float* __restrict__ h,
                                const float* __restrict__ Wc,
                                const float* __restrict__ bc,
                                float* __restrict__ out) {
    __shared__ float pooled[BQ][EQ];
    int t = threadIdx.x;
    for (int i = t; i < BQ * EQ; i += blockDim.x) {
        int b = i >> 7, e = i & 127;
        float s = 0.0f;
        for (int sidx = 0; sidx < SQ; ++sidx)
            s += h[((size_t)(b * SQ + sidx)) * EQ + e];
        pooled[b][e] = s * (1.0f / SQ);
    }
    __syncthreads();
    for (int i = t; i < BQ * CQ; i += blockDim.x) {
        int b = i / CQ, c = i % CQ;
        float s = bc[c];
        for (int e = 0; e < EQ; ++e) s += pooled[b][e] * Wc[c * EQ + e];
        out[b * CQ + c] = s;
    }
}

// ---------------------------------------------------------------------------
// Host launcher
// ---------------------------------------------------------------------------
extern "C" void kernel_launch(void* const* d_in, const int* in_sizes, int n_in,
                              void* d_out, int out_size, void* d_ws, size_t ws_size,
                              hipStream_t stream) {
    const int*   tokens     = (const int*)  d_in[0];
    const float* emb        = (const float*)d_in[1];
    const float* attn_theta = (const float*)d_in[2];   // (L, DK)
    const float* ffn_theta  = (const float*)d_in[3];   // (L, NQ)
    const float* Wo         = (const float*)d_in[4];   // (L, E, E)
    const float* W1         = (const float*)d_in[5];   // (L, F, NQ)
    const float* W2         = (const float*)d_in[6];   // (L, E, F)
    const float* ln1_g      = (const float*)d_in[7];
    const float* ln1_b      = (const float*)d_in[8];
    const float* ln2_g      = (const float*)d_in[9];
    const float* ln2_b      = (const float*)d_in[10];
    const float* Wc         = (const float*)d_in[11];
    const float* bc         = (const float*)d_in[12];
    float*       out        = (float*)d_out;

    // workspace layout (all f32): h | oattn | proj | a1   (~14 MB total)
    char* base = (char*)d_ws;
    float* h     = (float*)(base);
    float* oattn = (float*)(base + (size_t)BS * EQ * 4);
    float* proj  = (float*)(base + (size_t)BS * EQ * 4 * 2);
    float* a1    = (float*)(base + (size_t)BS * EQ * 4 * 3);

    qtc_embed_kernel<<<BS, EQ, 0, stream>>>(tokens, emb, h);

    for (int l = 0; l < LQ; ++l) {
        // --- quantum multi-head attention (flash-style, WMMA) ---
        qtc_attn_kernel<<<dim3(SQ / 16, HQ, BQ), 32, 0, stream>>>(
            h, attn_theta + l * DKQ, oattn);
        // output projection: oattn @ Wo^T  (4096 x 128 x 128)
        qtc_gemm_nt_wmma<<<dim3(BS / 16, EQ / 64), 32, 0, stream>>>(
            oattn, Wo + (size_t)l * EQ * EQ, proj, BS, EQ, EQ);
        qtc_add_ln_kernel<<<BS, 32, 0, stream>>>(
            h, proj, ln1_g + l * EQ, ln1_b + l * EQ);

        // --- quantum FFN ---
        qtc_ffn1_kernel<<<(BS * FQ + 255) / 256, 256, 0, stream>>>(
            h, ffn_theta + l * NQQ, W1 + (size_t)l * FQ * NQQ, a1);
        // a1 @ W2^T  (4096 x 128 x 512)
        qtc_gemm_nt_wmma<<<dim3(BS / 16, EQ / 64), 32, 0, stream>>>(
            a1, W2 + (size_t)l * EQ * FQ, proj, BS, EQ, FQ);
        qtc_add_ln_kernel<<<BS, 32, 0, stream>>>(
            h, proj, ln2_g + l * EQ, ln2_b + l * EQ);
    }

    qtc_head_kernel<<<1, 256, 0, stream>>>(h, Wc, bc, out);
}
